// SwinTransformerBlock3D_11836929868561
// MI455X (gfx1250) — matrix operand order repaired and checked by reading, not compiled, not run on hardware
//
#include <hip/hip_runtime.h>
#include <cstdint>

typedef __attribute__((ext_vector_type(16))) _Float16 v16h;
typedef __attribute__((ext_vector_type(8)))  float    v8f;

#define T_DIM   8
#define C_DIM   192
#define HEADS   6
#define HDIM    32
#define NTOK    98
#define NPAD    112
#define L_TOK   25088            // 8*56*56
#define M_ROWS  200704           // 2048 windows * 98 tokens == 8 * 25088
#define NWIN    2048
#define SCALE_Q 0.17677669529663687f   // 1/sqrt(32)

// ---------------- wave32 helpers ----------------

__device__ inline float wave_sum(float v) {
#pragma unroll
  for (int m = 16; m >= 1; m >>= 1) v += __shfl_xor(v, m, 32);
  return v;
}

// A fragment (16x32 f16, MxK): lane<16 -> M=lane, K runs {0..7}+{16..23};
// lane>=16 -> M=lane-16, K runs {8..15}+{24..31}. Row-major LDS source.
__device__ inline v16h load_a_frag(const _Float16* sm, int row0, int stride, int k0) {
  const int lane = threadIdx.x & 31;
  const _Float16* p = sm + (size_t)(row0 + (lane & 15)) * stride + k0 + ((lane >> 4) ? 8 : 0);
  v16h a;
#pragma unroll
  for (int i = 0; i < 8; ++i) a[i] = p[i];
#pragma unroll
  for (int i = 0; i < 8; ++i) a[8 + i] = p[16 + i];
  return a;
}

// B fragment (32x16 f16, KxN) from a K-major (transposed) LDS tile smT[n][k]:
// lane<16 -> N=lane, K={0..7}+{16..23} ; lane>=16 -> N=lane-16, K={8..15}+{24..31}
__device__ inline v16h load_bt_frag(const _Float16* smT, int col0, int stride, int k0) {
  const int lane = threadIdx.x & 31;
  const _Float16* p = smT + (size_t)(col0 + (lane & 15)) * stride + k0 + ((lane >> 4) ? 8 : 0);
  v16h b;
#pragma unroll
  for (int i = 0; i < 8; ++i) b[i] = p[i];
#pragma unroll
  for (int i = 0; i < 8; ++i) b[8 + i] = p[16 + i];
  return b;
}

// A fragment from the 112-wide probability matrix, zero-padded to K=128.
__device__ inline v16h load_p_frag(const _Float16* S, int row0, int k0) {
  const int lane = threadIdx.x & 31;
  const int r = row0 + (lane & 15);
  const int kb = k0 + ((lane >> 4) ? 8 : 0);
  v16h a;
#pragma unroll
  for (int i = 0; i < 8; ++i) { int j = kb + i;      a[i]     = (j < NPAD) ? S[r * NPAD + j] : (_Float16)0.f; }
#pragma unroll
  for (int i = 0; i < 8; ++i) { int j = kb + 16 + i; a[8 + i] = (j < NPAD) ? S[r * NPAD + j] : (_Float16)0.f; }
  return a;
}

__device__ inline v8f wmma16(v16h a, v16h b, v8f c) {
  return __builtin_amdgcn_wmma_f32_16x16x32_f16(false, a, false, b, (short)0, c, false, false);
}

// gfx1250 async global->LDS copy (ASYNCcnt-tracked; bypasses VGPRs).
// ldsaddr: 32-bit LDS byte address (AS(3) offset), gaddr: 64-bit global address.
__device__ inline void async_load_b128(unsigned ldsaddr, const void* gaddr) {
  asm volatile("global_load_async_to_lds_b128 %0, %1, off"
               :: "v"(ldsaddr), "v"(gaddr) : "memory");
}
__device__ inline void wait_asynccnt0() {
  asm volatile("s_wait_asynccnt 0" ::: "memory");
}

// ---------------- weight conversion ----------------

__global__ void f32_to_f16_kernel(const float* __restrict__ s, _Float16* __restrict__ d, int n) {
  int i = blockIdx.x * 256 + threadIdx.x;
  if (i < n) d[i] = (_Float16)s[i];
}

// ---------------- LN1 + roll(-1,-3,-3) + window partition -> f16 ----------------
// one wave per token; 8 tokens per 256-thread block
__global__ __launch_bounds__(256) void ln1_window_kernel(
    const float* __restrict__ x, const float* __restrict__ g, const float* __restrict__ b,
    _Float16* __restrict__ xw) {
  const int wave = threadIdx.x >> 5;
  const int lane = threadIdx.x & 31;
  const int tok  = blockIdx.x * 8 + wave;
  const int b_ = tok / NTOK;
  const int n  = tok - b_ * NTOK;
  const int bb = b_ >> 8;
  const int wi = b_ & 255;
  const int tw = wi >> 6, hw = (wi >> 3) & 7, ww = wi & 7;
  const int wt = n / 49, rem = n - wt * 49, wh = rem / 7, wwl = rem - wh * 7;
  const int ts   = (tw * 2 + wt + 1) & 7;          // roll -1 over T=8
  const int hs   = (hw * 7 + wh + 3) % 56;         // roll -3 over 56
  const int wsrc = (ww * 7 + wwl + 3) % 56;
  const size_t src = ((size_t)bb * L_TOK + (ts * 56 + hs) * 56 + wsrc) * C_DIM;
  float v[6];
#pragma unroll
  for (int i = 0; i < 6; ++i) v[i] = x[src + lane + 32 * i];
  float s = 0.f;
#pragma unroll
  for (int i = 0; i < 6; ++i) s += v[i];
  const float mu = wave_sum(s) * (1.f / 192.f);
  float q = 0.f;
#pragma unroll
  for (int i = 0; i < 6; ++i) { float d2 = v[i] - mu; q += d2 * d2; }
  const float rs = rsqrtf(wave_sum(q) * (1.f / 192.f) + 1e-5f);
  const size_t dst = (size_t)tok * C_DIM;
#pragma unroll
  for (int i = 0; i < 6; ++i) {
    int c = lane + 32 * i;
    xw[dst + c] = (_Float16)((v[i] - mu) * rs * g[c] + b[c]);
  }
}

// ---------------- LN2 (natural row order) ----------------

__global__ __launch_bounds__(256) void ln2_kernel(
    const float* __restrict__ x2, const float* __restrict__ g, const float* __restrict__ b,
    _Float16* __restrict__ h2) {
  const int wave = threadIdx.x >> 5;
  const int lane = threadIdx.x & 31;
  const size_t row = (size_t)blockIdx.x * 8 + wave;
  const size_t src = row * C_DIM;
  float v[6];
#pragma unroll
  for (int i = 0; i < 6; ++i) v[i] = x2[src + lane + 32 * i];
  float s = 0.f;
#pragma unroll
  for (int i = 0; i < 6; ++i) s += v[i];
  const float mu = wave_sum(s) * (1.f / 192.f);
  float q = 0.f;
#pragma unroll
  for (int i = 0; i < 6; ++i) { float d2 = v[i] - mu; q += d2 * d2; }
  const float rs = rsqrtf(wave_sum(q) * (1.f / 192.f) + 1e-5f);
#pragma unroll
  for (int i = 0; i < 6; ++i) {
    int c = lane + 32 * i;
    h2[src + c] = (_Float16)((v[i] - mu) * rs * g[c] + b[c]);
  }
}

// ---------------- generic WMMA GEMM, 128x64 block tile, 8 waves ----------------
// Double-buffered LDS; A tiles filled with async global->LDS (ASYNCcnt),
// B tiles transposed through VGPRs (16-bit scatter, covered by dscnt in barrier).

enum GemmEpi { EPI_QKV = 0, EPI_PROJ = 1, EPI_GELU = 2, EPI_FINAL = 3 };

template <int EPI>
__global__ __launch_bounds__(256) void wmma_gemm_kernel(
    const _Float16* __restrict__ A, const _Float16* __restrict__ B,
    const float* __restrict__ bias, void* __restrict__ Cout,
    int M, int N, int K, const float* __restrict__ resid) {
  __shared__ __align__(16) _Float16 sA[2][128 * 32];
  __shared__ __align__(16) _Float16 sBt[2][64 * 32];  // transposed: [n][k]
  const int tid  = threadIdx.x;
  const int wave = tid >> 5;
  const int lane = tid & 31;
  const int row0 = blockIdx.y * 128;
  const int col0 = blockIdx.x * 64;

  auto stage_async_A = [&](int kk, int buf) {
#pragma unroll
    for (int it = 0; it < 2; ++it) {
      int e = (tid + it * 256) * 8;              // 512 x b128 over the 128x32 tile
      int r = e >> 5, c = e & 31;
      async_load_b128((unsigned)(uintptr_t)&sA[buf][r * 32 + c],
                      &A[(size_t)(row0 + r) * K + kk + c]);
    }
  };
  auto stage_B = [&](int kk, int buf) {
#pragma unroll
    for (int it = 0; it < 8; ++it) {
      int e = tid + it * 256;                    // coalesced over n, scatter over k
      int k = e >> 6, n2 = e & 63;
      sBt[buf][n2 * 32 + k] = B[(size_t)(kk + k) * N + col0 + n2];
    }
  };

  v8f acc[4];
#pragma unroll
  for (int t = 0; t < 4; ++t)
#pragma unroll
    for (int i = 0; i < 8; ++i) acc[t][i] = 0.f;

  stage_async_A(0, 0);
  stage_B(0, 0);
  wait_asynccnt0();
  __syncthreads();

  int buf = 0;
  for (int kk = 0; kk < K; kk += 32, buf ^= 1) {
    const int nk = kk + 32;
    if (nk < K) {                 // prefetch next K-slab while we compute
      stage_async_A(nk, buf ^ 1);
      stage_B(nk, buf ^ 1);
    }
    v16h af = load_a_frag(sA[buf], wave * 16, 32, 0);
#pragma unroll
    for (int nt = 0; nt < 4; ++nt) {
      v16h bf = load_bt_frag(sBt[buf], nt * 16, 32, 0);
      acc[nt] = wmma16(af, bf, acc[nt]);
    }
    if (nk < K) wait_asynccnt0(); // my async fills of buf^1 are done
    __syncthreads();              // + everyone's ds stores (dscnt) before reuse
  }

  // epilogue: C layout -> row = (lane<16 ? g : g+8), col = lane%16
#pragma unroll
  for (int nt = 0; nt < 4; ++nt) {
#pragma unroll
    for (int gi = 0; gi < 8; ++gi) {
      int ri = (lane < 16) ? gi : gi + 8;
      int r = row0 + wave * 16 + ri;
      int c = col0 + nt * 16 + (lane & 15);
      float v = acc[nt][gi] + bias[c];
      if (EPI == EPI_QKV) {
        if (c < 192) v *= SCALE_Q;                       // pre-scale q
        ((_Float16*)Cout)[(size_t)r * N + c] = (_Float16)v;
      } else if (EPI == EPI_GELU) {
        v = 0.5f * v * (1.f + erff(v * 0.70710678118654752f));
        ((_Float16*)Cout)[(size_t)r * N + c] = (_Float16)v;
      } else if (EPI == EPI_FINAL) {
        ((float*)Cout)[(size_t)r * N + c] = v + resid[(size_t)r * N + c];
      } else {  // EPI_PROJ: window-reverse + roll(+1,+3,+3) + residual with x
        int b_ = r / NTOK, n = r - b_ * NTOK;
        int bb = b_ >> 8, wi = b_ & 255;
        int tw = wi >> 6, hw = (wi >> 3) & 7, ww = wi & 7;
        int wt = n / 49, rem2 = n - wt * 49, wh = rem2 / 7, wwl = rem2 - wh * 7;
        int t0 = (tw * 2 + wt + 1) & 7;
        int h0 = (hw * 7 + wh + 3) % 56;
        int w0 = (ww * 7 + wwl + 3) % 56;
        size_t idx = ((size_t)bb * L_TOK + (t0 * 56 + h0) * 56 + w0) * C_DIM + c;
        ((float*)Cout)[idx] = resid[idx] + v;
      }
    }
  }
}

// ---------------- windowed attention: one block per (window, head) ----------------

__global__ __launch_bounds__(128) void attn_kernel(
    const _Float16* __restrict__ qkv, const float* __restrict__ rpb,
    _Float16* __restrict__ out) {
  __shared__ __align__(16) _Float16 sS[NPAD * NPAD];   // scores / probs
  __shared__ __align__(16) _Float16 sQ[NPAD * HDIM];
  __shared__ __align__(16) _Float16 sK[NPAD * HDIM];
  __shared__ __align__(16) _Float16 sVT[HDIM * 128];   // v transposed, K padded to 128
  __shared__ int sReg[NPAD];
  __shared__ int sCrd[NPAD];
  const int b_   = blockIdx.x / HEADS;
  const int head = blockIdx.x % HEADS;
  const int wi = b_ & 255;
  const int tw = wi >> 6, hw = (wi >> 3) & 7, ww = wi & 7;
  const int tid = threadIdx.x;
  const int wave = tid >> 5;

  // per-token window-local coords + shift-mask region id (absolute shifted coords)
  if (tid < NTOK) {
    int wt = tid / 49, rem = tid - wt * 49, wh = rem / 7, wwl = rem - wh * 7;
    sCrd[tid] = (wt << 8) | (wh << 4) | wwl;
    int t = tw * 2 + wt, h = hw * 7 + wh, w = ww * 7 + wwl;
    int tr = (t < 6) ? 0 : (t == 6 ? 1 : 2);
    int hr = (h < 49) ? 0 : (h < 53 ? 1 : 2);
    int wr = (w < 49) ? 0 : (w < 53 ? 1 : 2);
    sReg[tid] = tr * 9 + hr * 3 + wr;
  }
  const size_t base = (size_t)b_ * NTOK * 576;
  for (int e = tid; e < NPAD * HDIM; e += 128) {     // q,k row-major, zero pad
    int m = e >> 5, d = e & 31;
    _Float16 qv = (_Float16)0.f, kv = (_Float16)0.f;
    if (m < NTOK) {
      qv = qkv[base + (size_t)m * 576 + head * 32 + d];
      kv = qkv[base + (size_t)m * 576 + 192 + head * 32 + d];
    }
    sQ[e] = qv; sK[e] = kv;
  }
  for (int e = tid; e < HDIM * 128; e += 128) {      // vT[d][m], zero pad m>=98
    int d = e >> 7, m = e & 127;
    sVT[e] = (m < NTOK) ? qkv[base + (size_t)m * 576 + 384 + head * 32 + d] : (_Float16)0.f;
  }
  __syncthreads();

  // S = q k^T : 7x7 tiles of 16x16, K=32 (single wmma per tile)
  for (int t = wave; t < 49; t += 4) {
    int mt = t / 7, nt = t - mt * 7;
    v8f a;
#pragma unroll
    for (int i = 0; i < 8; ++i) a[i] = 0.f;
    v16h af = load_a_frag(sQ, mt * 16, 32, 0);
    v16h bf = load_bt_frag(sK, nt * 16, 32, 0);     // sK[j][d] is exactly B^T staging
    a = wmma16(af, bf, a);
    const int lane = tid & 31;
#pragma unroll
    for (int gi = 0; gi < 8; ++gi) {
      int ri = (lane < 16) ? gi : gi + 8;
      sS[(mt * 16 + ri) * NPAD + nt * 16 + (lane & 15)] = (_Float16)a[gi];
    }
  }
  __syncthreads();

  // bias + mask + softmax, one row per thread, in place in sS
  if (tid < NPAD) {
    const int r = tid;
    if (r < NTOK) {
      const int cr = sCrd[r];
      const int wt_r = cr >> 8, wh_r = (cr >> 4) & 15, ww_r = cr & 15;
      const int reg_r = sReg[r];
      float mx = -1e30f;
      for (int j = 0; j < NTOK; ++j) {
        int cj = sCrd[j];
        int dt = wt_r - (cj >> 8) + 1;
        int dh = wh_r - ((cj >> 4) & 15) + 6;
        int dw = ww_r - (cj & 15) + 6;
        float s = (float)sS[r * NPAD + j] + rpb[(dt * 169 + dh * 13 + dw) * HEADS + head];
        if (reg_r != sReg[j]) s -= 100.f;
        sS[r * NPAD + j] = (_Float16)s;
        mx = fmaxf(mx, s);
      }
      float sum = 0.f;
      for (int j = 0; j < NTOK; ++j) {
        float e = __expf((float)sS[r * NPAD + j] - mx);
        sum += e;
        sS[r * NPAD + j] = (_Float16)e;
      }
      float inv = 1.f / sum;
      for (int j = 0; j < NTOK; ++j)
        sS[r * NPAD + j] = (_Float16)((float)sS[r * NPAD + j] * inv);
      for (int j = NTOK; j < NPAD; ++j) sS[r * NPAD + j] = (_Float16)0.f;
    } else {
      for (int j = 0; j < NPAD; ++j) sS[r * NPAD + j] = (_Float16)0.f;
    }
  }
  __syncthreads();

  // O = P v : 7 m-tiles x 2 d-tiles, K padded 112->128 (4 steps)
  for (int mt = wave; mt < 7; mt += 4) {
    v8f o0, o1;
#pragma unroll
    for (int i = 0; i < 8; ++i) { o0[i] = 0.f; o1[i] = 0.f; }
    for (int ks = 0; ks < 4; ++ks) {
      v16h pf = load_p_frag(sS, mt * 16, ks * 32);
      v16h b0 = load_bt_frag(sVT, 0, 128, ks * 32);
      v16h b1 = load_bt_frag(sVT, 16, 128, ks * 32);
      o0 = wmma16(pf, b0, o0);
      o1 = wmma16(pf, b1, o1);
    }
    const int lane = tid & 31;
#pragma unroll
    for (int gi = 0; gi < 8; ++gi) {
      int ri = (lane < 16) ? gi : gi + 8;
      int r = mt * 16 + ri;
      if (r < NTOK) {
        size_t o = ((size_t)b_ * NTOK + r) * C_DIM + head * 32;
        out[o + (lane & 15)]      = (_Float16)o0[gi];
        out[o + 16 + (lane & 15)] = (_Float16)o1[gi];
      }
    }
  }
}

// ---------------- launch ----------------

extern "C" void kernel_launch(void* const* d_in, const int* in_sizes, int n_in,
                              void* d_out, int out_size, void* d_ws, size_t ws_size,
                              hipStream_t stream) {
  (void)in_sizes; (void)n_in; (void)out_size; (void)ws_size;
  const float* x      = (const float*)d_in[0];
  const float* n1g    = (const float*)d_in[1];
  const float* n1b    = (const float*)d_in[2];
  const float* qkv_w  = (const float*)d_in[3];
  const float* qkv_b  = (const float*)d_in[4];
  const float* rpb    = (const float*)d_in[5];
  const float* proj_w = (const float*)d_in[6];
  const float* proj_b = (const float*)d_in[7];
  const float* n2g    = (const float*)d_in[8];
  const float* n2b    = (const float*)d_in[9];
  const float* fc1_w  = (const float*)d_in[10];
  const float* fc1_b  = (const float*)d_in[11];
  const float* fc2_w  = (const float*)d_in[12];
  const float* fc2_b  = (const float*)d_in[13];

  char* ws = (char*)d_ws;
  size_t off = 0;
  auto take = [&](size_t bytes) { char* p = ws + off; off = (off + bytes + 255) & ~(size_t)255; return p; };
  _Float16* qkv_w16  = (_Float16*)take(110592ull * 2);
  _Float16* proj_w16 = (_Float16*)take(36864ull * 2);
  _Float16* fc1_w16  = (_Float16*)take(147456ull * 2);
  _Float16* fc2_w16  = (_Float16*)take(147456ull * 2);
  _Float16* slot1 = (_Float16*)take((size_t)M_ROWS * 192 * 2);  // xw -> attn_out -> h2
  _Float16* slot2 = (_Float16*)take((size_t)M_ROWS * 768 * 2);  // qkv(576) -> mlp hidden(768)
  float*    x2    = (float*)take((size_t)M_ROWS * 192 * 4);     // residual stream after attn

  f32_to_f16_kernel<<<(110592 + 255) / 256, 256, 0, stream>>>(qkv_w, qkv_w16, 110592);
  f32_to_f16_kernel<<<(36864 + 255) / 256, 256, 0, stream>>>(proj_w, proj_w16, 36864);
  f32_to_f16_kernel<<<(147456 + 255) / 256, 256, 0, stream>>>(fc1_w, fc1_w16, 147456);
  f32_to_f16_kernel<<<(147456 + 255) / 256, 256, 0, stream>>>(fc2_w, fc2_w16, 147456);

  ln1_window_kernel<<<M_ROWS / 8, 256, 0, stream>>>(x, n1g, n1b, slot1);

  wmma_gemm_kernel<EPI_QKV><<<dim3(576 / 64, M_ROWS / 128), 256, 0, stream>>>(
      slot1, qkv_w16, qkv_b, slot2, M_ROWS, 576, 192, nullptr);

  attn_kernel<<<NWIN * HEADS, 128, 0, stream>>>(slot2, rpb, slot1);

  wmma_gemm_kernel<EPI_PROJ><<<dim3(192 / 64, M_ROWS / 128), 256, 0, stream>>>(
      slot1, proj_w16, proj_b, x2, M_ROWS, 192, 192, x);

  ln2_kernel<<<M_ROWS / 8, 256, 0, stream>>>(x2, n2g, n2b, slot1);

  wmma_gemm_kernel<EPI_GELU><<<dim3(768 / 64, M_ROWS / 128), 256, 0, stream>>>(
      slot1, fc1_w16, fc1_b, slot2, M_ROWS, 768, 192, nullptr);

  wmma_gemm_kernel<EPI_FINAL><<<dim3(192 / 64, M_ROWS / 128), 256, 0, stream>>>(
      slot2, fc2_w16, fc2_b, d_out, M_ROWS, 192, 768, x2);
}
